// GraphSage_4922032521470
// MI455X (gfx1250) — compile-verified, matrix operand order checked by reading, not run on hardware
//
#include <hip/hip_runtime.h>
#include <hip/hip_bf16.h>

// GraphSAGE 2-layer forward for MI455X (gfx1250, wave32).
//   h' = tanh( [h | (sum_{e:dst=v} h[src_e] + h_v) / (deg_v + 1)] @ W^T )
// v2: W pre-converted to bf16 once (64KB, WGP$-resident); each wave computes a
// full 16x128 output strip (8 column tiles) so the A fragment is loaded and
// f32->bf16 converted once per 64 WMMAs instead of once per 8.

typedef __attribute__((ext_vector_type(16))) __bf16 v16bf;
typedef __attribute__((ext_vector_type(8)))  __bf16 bf16x8;
typedef __attribute__((ext_vector_type(8)))  float  v8f;

#define DFEAT 128
#define KDIM  256   // 2*DFEAT

// ---------------- graph-degree kernels ----------------

__global__ void zero_f32_kernel(float* __restrict__ p, int n) {
    int i = blockIdx.x * blockDim.x + threadIdx.x;
    if (i < n) p[i] = 0.0f;
}

__global__ void deg_accum_kernel(const int* __restrict__ dst, float* __restrict__ deg, int n_edges) {
    int e = blockIdx.x * blockDim.x + threadIdx.x;
    if (e < n_edges) atomicAdd(&deg[dst[e]], 1.0f);
}

__global__ void invert_deg_kernel(float* __restrict__ deg, int n) {
    int i = blockIdx.x * blockDim.x + threadIdx.x;
    if (i < n) deg[i] = 1.0f / (deg[i] + 1.0f);
}

// W (f32) -> bf16, once per layer weight
__global__ void conv_w_bf16_kernel(const float* __restrict__ W, __bf16* __restrict__ Wb, int n) {
    int i = blockIdx.x * blockDim.x + threadIdx.x;
    if (i < n) Wb[i] = (__bf16)W[i];
}

// ---------------- per-layer data movement ----------------

// sum = h  (float4 vectorized; D=128 -> rows are 16B aligned)
__global__ void copy_f32x4_kernel(const float* __restrict__ src, float* __restrict__ dstp, int n_vec4) {
    int i = blockIdx.x * blockDim.x + threadIdx.x;
    if (i < n_vec4) ((float4*)dstp)[i] = ((const float4*)src)[i];
}

// One wave per edge: lane l handles features 4l..4l+3 (32 lanes * 4 = 128).
// Emits no-return global_atomic_add_f32 (STOREcnt path), resolved in L2.
__global__ void scatter_add_kernel(const float* __restrict__ h,
                                   const int* __restrict__ src,
                                   const int* __restrict__ dst,
                                   float* __restrict__ sum, int n_edges) {
    int wave = blockIdx.x * (blockDim.x >> 5) + (threadIdx.x >> 5);
    int lane = threadIdx.x & 31;
    if (wave >= n_edges) return;
    int s = src[wave];
    int d = dst[wave];
    const float4 v = *(const float4*)(h + (size_t)s * DFEAT + lane * 4);
    float* p = sum + (size_t)d * DFEAT + lane * 4;
    atomicAdd(p + 0, v.x);
    atomicAdd(p + 1, v.y);
    atomicAdd(p + 2, v.z);
    atomicAdd(p + 3, v.w);
}

// ---------------- fused GEMM + tanh (bf16 WMMA) ----------------

// Load 8 consecutive f32, scale, convert to bf16 into dst8[0..7].
__device__ __forceinline__ void load8_bf16(const float* __restrict__ p, float scale, __bf16* dst8) {
    float4 x = *(const float4*)(p);
    float4 y = *(const float4*)(p + 4);
    dst8[0] = (__bf16)(x.x * scale);
    dst8[1] = (__bf16)(x.y * scale);
    dst8[2] = (__bf16)(x.z * scale);
    dst8[3] = (__bf16)(x.w * scale);
    dst8[4] = (__bf16)(y.x * scale);
    dst8[5] = (__bf16)(y.y * scale);
    dst8[6] = (__bf16)(y.z * scale);
    dst8[7] = (__bf16)(y.w * scale);
}

// out[n_nodes,128] = tanh( [h | sum*inv] @ Wb^T ),  Wb: [128, 256] bf16.
// 256 threads = 8 waves/block; each wave owns one 16-row strip and ALL 8
// column tiles: per K-step, one A fragment feeds 8 v_wmma (64 WMMA/wave).
// K step 4 is exactly the h|agg boundary, so each step has one source+scale.
__global__ void __launch_bounds__(256)
sage_gemm_tanh_kernel(const float* __restrict__ h,
                      const float* __restrict__ sum,
                      const float* __restrict__ invdeg,
                      const __bf16* __restrict__ Wb,
                      float* __restrict__ out, int n_nodes) {
    const int lane  = threadIdx.x & 31;
    const int wave  = threadIdx.x >> 5;
    const int rtile = blockIdx.x * 8 + wave;
    if (rtile * 16 >= n_nodes) return;

    const int m  = lane & 15;
    const int hi = lane >> 4;

    int row = rtile * 16 + m;
    if (row >= n_nodes) row = n_nodes - 1;        // N%16==0 here; safety clamp
    const float inv = invdeg[row];

    const float* __restrict__ hrow = h   + (size_t)row * DFEAT;
    const float* __restrict__ srow = sum + (size_t)row * DFEAT;

    v8f c[8];
#pragma unroll
    for (int ct = 0; ct < 8; ++ct) c[ct] = (v8f){};

#pragma unroll
    for (int ks = 0; ks < 8; ++ks) {
        const bool  aggp  = (ks >= 4);
        const float scale = aggp ? inv : 1.0f;
        const float* __restrict__ srcA = aggp ? srow : hrow;
        const int kbase = (ks & 3) * 32;          // within h or agg half

        // A fragment (16x32 bf16): lane(hi,m) holds K = kbase+hi*8..+7 and +16..
        v16bf a;
        load8_bf16(srcA + kbase + hi * 8,      scale, (__bf16*)&a);
        load8_bf16(srcA + kbase + 16 + hi * 8, scale, ((__bf16*)&a) + 8);

        const int kw = ks * 32;
#pragma unroll
        for (int ct = 0; ct < 8; ++ct) {
            // B fragment (32x16 bf16): lane&15 = column within tile, K striped as A.
            // Wb row n = ct*16+m, two 16B bf16 loads, zero VALU.
            const __bf16* __restrict__ wp = Wb + (size_t)(ct * 16 + m) * KDIM + kw + hi * 8;
            v16bf b;
            *((bf16x8*)&b)       = *(const bf16x8*)(wp);
            *(((bf16x8*)&b) + 1) = *(const bf16x8*)(wp + 16);

            c[ct] = __builtin_amdgcn_wmma_f32_16x16x32_bf16(
                /*neg_a=*/false, a, /*neg_b=*/false, b,
                /*c_mod=*/(short)0, c[ct], /*reuse_a=*/false, /*reuse_b=*/false);
        }
    }

    // C/D layout: VGPR r -> M = hi*8 + r; N = lane&15
#pragma unroll
    for (int ct = 0; ct < 8; ++ct) {
#pragma unroll
        for (int r = 0; r < 8; ++r) {
            int orow = rtile * 16 + hi * 8 + r;
            if (orow < n_nodes)
                out[(size_t)orow * DFEAT + ct * 16 + m] = tanhf(c[ct][r]);
        }
    }
}

// ---------------- host launch ----------------

extern "C" void kernel_launch(void* const* d_in, const int* in_sizes, int n_in,
                              void* d_out, int out_size, void* d_ws, size_t ws_size,
                              hipStream_t stream) {
    const float* feats = (const float*)d_in[0];   // [N,128] f32
    const int*   src   = (const int*)d_in[1];     // [E] i32
    const int*   dst   = (const int*)d_in[2];     // [E] i32
    const float* W1    = (const float*)d_in[3];   // [128,256] f32
    const float* W2    = (const float*)d_in[4];   // [128,256] f32
    float*       out   = (float*)d_out;           // [N,128] f32

    const int N = in_sizes[0] / DFEAT;
    const int E = in_sizes[1];
    const int WN = DFEAT * KDIM;                  // 32768 weight elements

    // workspace: sum [N*128] f32 | h1 [N*128] f32 | invdeg [N] f32 | Wb1,Wb2 bf16
    float* sumb = (float*)d_ws;
    float* h1   = sumb + (size_t)N * DFEAT;
    float* inv  = h1   + (size_t)N * DFEAT;
    size_t wboff = (((size_t)(inv + N) - (size_t)d_ws) + 255) & ~(size_t)255; // 256B align
    __bf16* Wb1 = (__bf16*)((char*)d_ws + wboff);
    __bf16* Wb2 = Wb1 + WN;

    const int T = 256;
    const int nvec4 = N * (DFEAT / 4);
    const int gCopy = (nvec4 + T - 1) / T;
    const int gNode = (N + T - 1) / T;
    const int gEdge = (E + T - 1) / T;
    const int gScat = (E + 7) / 8;                // 8 edges (waves) per 256-thr block
    const int gW    = (WN + T - 1) / T;
    const int gGemm = ((N + 15) / 16 + 7) / 8;    // 8 row-strip waves per block

    // one-time prep: degree -> 1/(deg+1); W1,W2 -> bf16
    zero_f32_kernel<<<gNode, T, 0, stream>>>(inv, N);
    deg_accum_kernel<<<gEdge, T, 0, stream>>>(dst, inv, E);
    invert_deg_kernel<<<gNode, T, 0, stream>>>(inv, N);
    conv_w_bf16_kernel<<<gW, T, 0, stream>>>(W1, Wb1, WN);
    conv_w_bf16_kernel<<<gW, T, 0, stream>>>(W2, Wb2, WN);

    // ---- layer 1 ----
    copy_f32x4_kernel<<<gCopy, T, 0, stream>>>(feats, sumb, nvec4);
    scatter_add_kernel<<<gScat, T, 0, stream>>>(feats, src, dst, sumb, E);
    sage_gemm_tanh_kernel<<<gGemm, T, 0, stream>>>(feats, sumb, inv, Wb1, h1, N);

    // ---- layer 2 ----
    copy_f32x4_kernel<<<gCopy, T, 0, stream>>>(h1, sumb, nvec4);
    scatter_add_kernel<<<gScat, T, 0, stream>>>(h1, src, dst, sumb, E);
    sage_gemm_tanh_kernel<<<gGemm, T, 0, stream>>>(h1, sumb, inv, Wb2, out, N);
}
